// NNConv_87436944212625
// MI455X (gfx1250) — matrix operand order repaired and checked by reading, not compiled, not run on hardware
//
#include <hip/hip_runtime.h>

#define NUM_NODES 50000
#define NUM_EDGES 300000
#define WIDTH 32
#define EDGE_DIM 6
#define HIDDEN 128
#define NTILES (NUM_EDGES / 16) // 18750
#define NPAIRS (NTILES / 2)     // 9375, exact

typedef __attribute__((ext_vector_type(16))) __bf16 v16bf;
typedef __attribute__((ext_vector_type(8))) float v8f;

union FragB32 {
    uint4 q[2];
    v16bf v;
};

__device__ __forceinline__ unsigned short f32_to_bf16(float f) {
    unsigned u = __float_as_uint(f);
    u += 0x7FFFu + ((u >> 16) & 1u); // round to nearest even
    return (unsigned short)(u >> 16);
}

// ---------------------------------------------------------------- zero output
__global__ void zero_f32(float* __restrict__ p, int n) {
    int i = blockIdx.x * blockDim.x + threadIdx.x;
    if (i < n) p[i] = 0.0f;
}

// ------------------------------------------- pre-swizzle W2 into B fragments
// B-fragment order for v_wmma_f32_16x16x32_bf16:
//   fragment id f = j*4 + s   (j = 16-col chunk 0..63, s = K-step 0..3)
//   lane l, element e (0..15):  K = s*32 + 8*(l/16) + e (e<8), s*32 + 16 + 8*(l/16) + (e-8) (e>=8),  N = j*16 + (l%16)
__global__ void prep_w2(const float* __restrict__ W2, unsigned short* __restrict__ frag) {
    int t = blockIdx.x * blockDim.x + threadIdx.x; // 131072 elements
    if (t >= HIDDEN * WIDTH * WIDTH) return;
    int e = t & 15;
    int l = (t >> 4) & 31;
    int ts = t >> 9; // j*4 + s
    int s = ts & 3;
    int j = ts >> 2;
    int k = s * 32 + 8 * (l >> 4) + (e & 7) + ((e >> 3) << 4);
    int n = j * 16 + (l & 15);
    frag[t] = f32_to_bf16(W2[k * (WIDTH * WIDTH) + n]);
}

// ------------------------------------------------------------- fused main
// Each wave register-blocks TWO 16-edge tiles (32 edges) so every B fragment
// feeds two WMMAs (1 global_load_b128 : 1 v_wmma instead of 2 : 1).
#define WAVES_PER_BLOCK 4
#define LDS_PER_WAVE 13312
// per-wave LDS layout (bytes):
//   [0,4096)       x_j tile, f32 [32 edges][32]
//   [4096,12288)   h tile, bf16 [32 edges][128]
//   [12288,13056)  edge_attr tile, f32 [32][6]
//   [13056,13184)  sender idx [32]
//   [13184,13312)  receiver idx [32]

__global__ __launch_bounds__(128) void nnconv_main(
    const float* __restrict__ x, const int* __restrict__ senders,
    const int* __restrict__ receivers, const float* __restrict__ edge_attr,
    const float* __restrict__ W1, const float* __restrict__ b1,
    const unsigned short* __restrict__ w2frag, const float* __restrict__ b2,
    float* __restrict__ out) {
    __shared__ float b2_lds[WIDTH * WIDTH];
    __shared__ unsigned char wmem[WAVES_PER_BLOCK * LDS_PER_WAVE];

    const int lane = threadIdx.x & 31;
    const int wid = threadIdx.x >> 5;
    const int m = lane & 15;  // C/A row & N-lane
    const int hh = lane >> 4; // lane half

    // stage b2 once per block
    for (int i = threadIdx.x; i < WIDTH * WIDTH; i += blockDim.x) b2_lds[i] = b2[i];
    __syncthreads();

    // W1/b1 resident in registers: w1reg[d][c] = W1[d][c*32+lane]
    float w1reg[EDGE_DIM][4];
    float b1reg[4];
#pragma unroll
    for (int c = 0; c < 4; ++c) {
        b1reg[c] = b1[c * 32 + lane];
#pragma unroll
        for (int d = 0; d < EDGE_DIM; ++d) w1reg[d][c] = W1[d * HIDDEN + c * 32 + lane];
    }

    float* x_lds = (float*)(wmem + wid * LDS_PER_WAVE);
    unsigned short* h_lds = (unsigned short*)(wmem + wid * LDS_PER_WAVE + 4096);
    float* ea_lds = (float*)(wmem + wid * LDS_PER_WAVE + 12288);
    int* sidx = (int*)(wmem + wid * LDS_PER_WAVE + 13056);
    int* ridx = (int*)(wmem + wid * LDS_PER_WAVE + 13184);

    const uint4* w2q = (const uint4*)w2frag;
    const int nwaves = gridDim.x * WAVES_PER_BLOCK;

    for (int pair = blockIdx.x * WAVES_PER_BLOCK + wid; pair < NPAIRS; pair += nwaves) {
        const int ebase = pair * 32;

        // ---- stage indices and edge attributes (wave-private LDS region)
        sidx[lane] = senders[ebase + lane];
        ridx[lane] = receivers[ebase + lane];
#pragma unroll
        for (int t = 0; t < 6; ++t)
            ea_lds[t * 32 + lane] = edge_attr[ebase * EDGE_DIM + t * 32 + lane];
        asm volatile("s_wait_dscnt 0" ::: "memory");

        // ---- gather x_j rows (coalesced per edge row)
#pragma unroll 4
        for (int r = 0; r < 32; ++r) {
            int sn = sidx[r];
            x_lds[r * WIDTH + lane] = x[sn * WIDTH + lane];
        }

        // ---- h = relu(ea @ W1 + b1) -> bf16 LDS [32][128]
        for (int r = 0; r < 32; ++r) {
            float e0 = ea_lds[r * 6 + 0], e1 = ea_lds[r * 6 + 1], e2 = ea_lds[r * 6 + 2];
            float e3 = ea_lds[r * 6 + 3], e4 = ea_lds[r * 6 + 4], e5 = ea_lds[r * 6 + 5];
#pragma unroll
            for (int c = 0; c < 4; ++c) {
                float acc = b1reg[c];
                acc = fmaf(e0, w1reg[0][c], acc);
                acc = fmaf(e1, w1reg[1][c], acc);
                acc = fmaf(e2, w1reg[2][c], acc);
                acc = fmaf(e3, w1reg[3][c], acc);
                acc = fmaf(e4, w1reg[4][c], acc);
                acc = fmaf(e5, w1reg[5][c], acc);
                acc = fmaxf(acc, 0.0f);
                h_lds[r * HIDDEN + c * 32 + lane] = f32_to_bf16(acc);
            }
        }
        asm volatile("s_wait_dscnt 0" ::: "memory");

        // ---- load A fragments for both tiles (resident for whole pair)
        // lane l: row M = l%16, elems e=0..7 -> K = s*32 + 8*(l/16) + e,
        //                     elems e=8..15 -> K = s*32 + 16 + 8*(l/16) + (e-8)
        FragB32 afrag[2][4];
        const uint4* hq = (const uint4*)h_lds;
#pragma unroll
        for (int t = 0; t < 2; ++t)
#pragma unroll
            for (int s = 0; s < 4; ++s) {
                int base = ((t * 16 + m) * HIDDEN + s * 32 + hh * 8) >> 3;
                afrag[t][s].q[0] = hq[base];
                afrag[t][s].q[1] = hq[base + 2];
            }

        float macc[2][2][8]; // [tile][half][row]
#pragma unroll
        for (int t = 0; t < 2; ++t)
#pragma unroll
            for (int hf = 0; hf < 2; ++hf)
#pragma unroll
                for (int r = 0; r < 8; ++r) macc[t][hf][r] = 0.0f;

        // ---- main loop: weight slabs via WMMA, immediate contraction with x_j
        for (int w = 0; w < WIDTH; ++w) {
            float xw[2][8];
#pragma unroll
            for (int t = 0; t < 2; ++t)
#pragma unroll
                for (int r = 0; r < 8; ++r)
                    xw[t][r] = x_lds[(t * 16 + r + 8 * hh) * WIDTH + w];
#pragma unroll
            for (int half = 0; half < 2; ++half) {
                int j = 2 * w + half; // N-chunk 0..63
                // prefetch an upcoming B slab toward the WGP
                if (j + 4 < 64) {
                    int pf = (((j + 4) << 2) * 32 + lane) * 2;
                    __builtin_prefetch(&w2q[pf], 0, 3);
                }
                float b2n = b2_lds[j * 16 + m];
                FragB32 bfrag[4];
#pragma unroll
                for (int s = 0; s < 4; ++s) {
                    int fb = (((j << 2) + s) * 32 + lane) * 2;
                    bfrag[s].q[0] = w2q[fb];
                    bfrag[s].q[1] = w2q[fb + 1];
                }
#pragma unroll
                for (int t = 0; t < 2; ++t) {
                    // fold b2 through the x-contraction: C init = b2[n]
                    v8f c;
#pragma unroll
                    for (int r = 0; r < 8; ++r) c[r] = b2n;
#pragma unroll
                    for (int s = 0; s < 4; ++s)
                        c = __builtin_amdgcn_wmma_f32_16x16x32_bf16(
                            false, afrag[t][s].v, false, bfrag[s].v, (short)0, c, false, false);
                    // C layout: VGPR r, lane l -> edge = r + 8*(l/16), v = half*16 + l%16
#pragma unroll
                    for (int r = 0; r < 8; ++r)
                        macc[t][half][r] = fmaf(c[r], xw[t][r], macc[t][half][r]);
                }
            }
        }

        // ---- scatter-add into receivers
#pragma unroll
        for (int t = 0; t < 2; ++t)
#pragma unroll
            for (int r = 0; r < 8; ++r) {
                int e = t * 16 + r + 8 * hh;
                int dst = ridx[e] * WIDTH;
                atomicAdd(&out[dst + m], macc[t][0][r]);
                atomicAdd(&out[dst + 16 + m], macc[t][1][r]);
            }
    }
}

// ------------------------------------------------------------------ launcher
extern "C" void kernel_launch(void* const* d_in, const int* in_sizes, int n_in,
                              void* d_out, int out_size, void* d_ws, size_t ws_size,
                              hipStream_t stream) {
    const float* x = (const float*)d_in[0];
    const int* senders = (const int*)d_in[1];
    const int* receivers = (const int*)d_in[2];
    const float* edge_attr = (const float*)d_in[3];
    const float* W1 = (const float*)d_in[4];
    const float* b1 = (const float*)d_in[5];
    const float* W2 = (const float*)d_in[6];
    const float* b2 = (const float*)d_in[7];
    float* out = (float*)d_out;
    unsigned short* w2frag = (unsigned short*)d_ws; // 256 KB

    zero_f32<<<(out_size + 255) / 256, 256, 0, stream>>>(out, out_size);
    prep_w2<<<(HIDDEN * WIDTH * WIDTH + 255) / 256, 256, 0, stream>>>(W2, w2frag);
    nnconv_main<<<2048, 128, 0, stream>>>(x, senders, receivers, edge_attr,
                                          W1, b1, w2frag, b2, out);
}
